// WindowAttention_v2_59373627899922
// MI455X (gfx1250) — matrix operand order repaired and checked by reading, not compiled, not run on hardware
//
#include <hip/hip_runtime.h>
#include <math.h>

// CDNA5 (gfx1250) wave32 WMMA types
typedef __attribute__((ext_vector_type(16))) _Float16 v16h;
typedef __attribute__((ext_vector_type(8)))  _Float16 v8h;
typedef __attribute__((ext_vector_type(8)))  float    v8f;

#define NH_     6
#define HD_     32          // head dim
#define NTOK    64          // tokens per window (8x8)
#define BATCH   2
#define CCH     192
#define PLANE   65536       // 256*256
#define LOGIT_MAX 4.605170185988092f   // log(100)
#define INV_SQRT_HD 0.17677669529663687f

// ---------------------------------------------------------------------------
// Kernel 1: continuous-position-bias MLP  bt[225][6] = relu(table@w1+b1)@w2
// ---------------------------------------------------------------------------
__global__ void cpb_mlp_kernel(const float* __restrict__ table,
                               const float* __restrict__ w1,   // (2,512)
                               const float* __restrict__ b1,   // (512,)
                               const float* __restrict__ w2,   // (512,6)
                               float* __restrict__ bt)         // (225,6)
{
    int gid = blockIdx.x * blockDim.x + threadIdx.x;
    if (gid >= 225 * NH_) return;
    int p = gid / NH_;
    int h = gid - p * NH_;
    float t0 = table[2 * p + 0];
    float t1 = table[2 * p + 1];
    float acc = 0.f;
    for (int k = 0; k < 512; ++k) {
        float hk = fmaf(t0, w1[k], fmaf(t1, w1[512 + k], b1[k]));
        hk = fmaxf(hk, 0.f);                       // relu
        acc = fmaf(hk, w2[k * NH_ + h], acc);
    }
    bt[p * NH_ + h] = acc;
}

// ---------------------------------------------------------------------------
// Kernel 2: gather + 16*sigmoid  bias[h][n][m] = 16*sigmoid(bt[index[n][m]][h])
// ---------------------------------------------------------------------------
__global__ void bias_gather_kernel(const float* __restrict__ bt,
                                   const int*   __restrict__ index,  // (64,64)
                                   float* __restrict__ bias)         // (6,64,64)
{
    int gid = blockIdx.x * blockDim.x + threadIdx.x;
    if (gid >= NH_ * NTOK * NTOK) return;
    int h  = gid / (NTOK * NTOK);
    int nm = gid - h * (NTOK * NTOK);
    int idx = index[nm];
    float x = bt[idx * NH_ + h];
    bias[gid] = 16.f / (1.f + __expf(-x));
}

// ---------------------------------------------------------------------------
// Kernel 3: windowed attention, one block (4 wave32) per window.
// ---------------------------------------------------------------------------
__global__ __launch_bounds__(128)
void win_attn_kernel(const float* __restrict__ qkv,          // (3,B,192,256,256)
                     const float* __restrict__ mask,         // (1024,64,64)
                     const float* __restrict__ logit_scale,  // (6,)
                     const float* __restrict__ bias,         // (6,64,64) from ws
                     float* __restrict__ out)                // (B,192,256,256)
{
    __shared__ __attribute__((aligned(32))) _Float16 Qs[NTOK * HD_];  // [tok][d]
    __shared__ __attribute__((aligned(32))) _Float16 Ks[NTOK * HD_];  // [tok][d]
    __shared__ __attribute__((aligned(32))) _Float16 Vt[HD_ * NTOK];  // [d][tok]
    __shared__ __attribute__((aligned(32))) float    Sm[NTOK * NTOK]; // S / O stage
    __shared__ __attribute__((aligned(32))) _Float16 Ps[NTOK * NTOK]; // softmax(S) f16

    const int tid  = threadIdx.x;
    const int lane = tid & 31;
    const int wave = tid >> 5;          // 0..3 : query tile
    const int lhi  = (lane >> 4) & 1;   // upper lane half
    const int ln16 = lane & 15;

    const int blk  = blockIdx.x;        // 0..2047
    const int b    = blk >> 10;
    const int wIdx = blk & 1023;        // window id within image (wy*32+wx)
    const int wy   = wIdx >> 5;
    const int wx   = wIdx & 31;

    const float* maskw = mask + (size_t)wIdx * (NTOK * NTOK);

    for (int h = 0; h < NH_; ++h) {
        __syncthreads();   // protect LDS reuse across head iterations

        // ---- stage Q,K,V (f32 -> f16) into LDS; gather applies roll(-4,-4) ----
        const float* qbase = qkv + ((size_t)(0 * BATCH + b) * CCH + h * HD_) * PLANE;
        const float* kbase = qkv + ((size_t)(1 * BATCH + b) * CCH + h * HD_) * PLANE;
        const float* vbase = qkv + ((size_t)(2 * BATCH + b) * CCH + h * HD_) * PLANE;
        for (int idx = tid; idx < NTOK * HD_; idx += 128) {
            int dim   = idx >> 6;        // 0..31
            int token = idx & 63;        // consecutive threads -> consecutive cols
            int i = token >> 3, j = token & 7;
            int hh = (wy * 8 + i + 4) & 255;
            int ww = (wx * 8 + j + 4) & 255;
            int po = dim * PLANE + (hh << 8) + ww;
            Qs[token * HD_ + dim]  = (_Float16)qbase[po];
            Ks[token * HD_ + dim]  = (_Float16)kbase[po];
            Vt[dim * NTOK + token] = (_Float16)vbase[po];
        }
        __syncthreads();

        const float sc = __expf(fminf(logit_scale[h], LOGIT_MAX)) * INV_SQRT_HD;
        const float* biash = bias + h * (NTOK * NTOK);

        // ---- GEMM1: S = Q*K^T  (M=64,N=64,K=32) -> 4x4 tiles, wave owns mtile ----
        const int mtile = wave;
        const int am    = mtile * 16 + ln16;      // A row = query (M = lane%16)
        const int koff  = lhi * 16;               // A layout: contiguous run [koff,+16)
        v8h  alo = *(const v8h*)&Qs[am * HD_ + koff];
        v8h  ahi = *(const v8h*)&Qs[am * HD_ + koff + 8];
        v16h Aq  = __builtin_shufflevector(alo, ahi,
                     0,1,2,3,4,5,6,7,8,9,10,11,12,13,14,15);

        for (int ntile = 0; ntile < 4; ++ntile) {
            int bn = ntile * 16 + ln16;           // B col = key (N = lane%16)
            // B layout: k = e + 16*(lane>=16), contiguous 16 halves (2x ds_read_b128)
            v16h Bk = *(const v16h*)&Ks[bn * HD_ + lhi * 16];
            v8f c = {};
            v8f d = __builtin_amdgcn_wmma_f32_16x16x32_f16(
                        false, Aq, false, Bk, (short)0, c, false, false);
            #pragma unroll
            for (int r = 0; r < 8; ++r) {         // C/D: M = r + 8*lhi, N = lane%16
                int qrow = mtile * 16 + r + 8 * lhi;
                int key  = ntile * 16 + ln16;
                int o = qrow * NTOK + key;
                Sm[o] = d[r] * sc + biash[o] + maskw[o];
            }
        }
        __syncthreads();

        // ---- softmax over keys (rows of Sm): 2 threads per row ----
        {
            int row   = tid >> 1;
            int halfo = (tid & 1) * 32;
            float* srow = &Sm[row * NTOK + halfo];
            float mx = -3.402823e38f;
            #pragma unroll
            for (int jj = 0; jj < 32; ++jj) mx = fmaxf(mx, srow[jj]);
            mx = fmaxf(mx, __shfl_xor(mx, 1));
            float sum = 0.f;
            #pragma unroll
            for (int jj = 0; jj < 32; ++jj) {
                float e = __expf(srow[jj] - mx);
                srow[jj] = e;
                sum += e;
            }
            sum += __shfl_xor(sum, 1);
            float inv = 1.f / sum;
            _Float16* prow = &Ps[row * NTOK + halfo];
            #pragma unroll
            for (int jj = 0; jj < 32; ++jj) prow[jj] = (_Float16)(srow[jj] * inv);
        }
        __syncthreads();

        // ---- GEMM2: O = P*V  (M=64,N=32,K=64) -> 4x2 tiles, 2 chained K=32 ----
        v16h Ap[2];
        #pragma unroll
        for (int kc = 0; kc < 2; ++kc) {
            v8h plo = *(const v8h*)&Ps[am * NTOK + kc * 32 + koff];
            v8h phi = *(const v8h*)&Ps[am * NTOK + kc * 32 + koff + 8];
            Ap[kc] = __builtin_shufflevector(plo, phi,
                       0,1,2,3,4,5,6,7,8,9,10,11,12,13,14,15);
        }
        #pragma unroll
        for (int ntile = 0; ntile < 2; ++ntile) {
            int nd = ntile * 16 + ln16;           // output head-dim index
            v8f acc = {};
            #pragma unroll
            for (int kc = 0; kc < 2; ++kc) {
                v16h Bv = *(const v16h*)&Vt[nd * NTOK + kc * 32 + lhi * 16];
                acc = __builtin_amdgcn_wmma_f32_16x16x32_f16(
                          false, Ap[kc], false, Bv, (short)0, acc, false, false);
            }
            #pragma unroll
            for (int r = 0; r < 8; ++r) {
                int qrow = mtile * 16 + r + 8 * lhi;
                Sm[qrow * HD_ + nd] = acc[r];     // stage O[tok][dim] (reuse Sm)
            }
        }
        __syncthreads();

        // ---- coalesced writeout with inverse roll (+4,+4) ----
        float* obase = out + ((size_t)b * CCH + h * HD_) * PLANE;
        for (int idx = tid; idx < NTOK * HD_; idx += 128) {
            int dim   = idx >> 6;
            int token = idx & 63;
            int i = token >> 3, j = token & 7;
            int hh = (wy * 8 + i + 4) & 255;
            int ww = (wx * 8 + j + 4) & 255;
            obase[dim * PLANE + (hh << 8) + ww] = Sm[token * HD_ + dim];
        }
    }
}

// ---------------------------------------------------------------------------
extern "C" void kernel_launch(void* const* d_in, const int* in_sizes, int n_in,
                              void* d_out, int out_size, void* d_ws, size_t ws_size,
                              hipStream_t stream)
{
    const float* qkv         = (const float*)d_in[0];   // (3,2,192,256,256)
    const float* table       = (const float*)d_in[1];   // (1,15,15,2)
    const int*   index       = (const int*)  d_in[2];   // (64,64)
    const float* mask        = (const float*)d_in[3];   // (1024,64,64)
    const float* logit_scale = (const float*)d_in[4];   // (6,1,1)
    const float* w1          = (const float*)d_in[5];   // (2,512)
    const float* b1          = (const float*)d_in[6];   // (512,)
    const float* w2          = (const float*)d_in[7];   // (512,6)
    float*       out         = (float*)d_out;

    // workspace: bt (225*6 floats) at 0, bias (6*64*64 floats) at float ofs 2048
    float* bt   = (float*)d_ws;
    float* bias = (float*)d_ws + 2048;

    cpb_mlp_kernel<<<(225 * NH_ + 255) / 256, 256, 0, stream>>>(table, w1, b1, w2, bt);
    bias_gather_kernel<<<(NH_ * NTOK * NTOK + 255) / 256, 256, 0, stream>>>(bt, index, bias);
    win_attn_kernel<<<2048, 128, 0, stream>>>(qkv, mask, logit_scale, bias, out);
}
